// MultiHeadAttention_43241730736684
// MI455X (gfx1250) — matrix operand order repaired and checked by reading, not compiled, not run on hardware
//
#include <hip/hip_runtime.h>
#include <hip/hip_bf16.h>
#include <cstdint>

typedef _Float16 v16h __attribute__((ext_vector_type(16)));
typedef _Float16 v8h  __attribute__((ext_vector_type(8)));
typedef float    v8f  __attribute__((ext_vector_type(8)));
typedef float    v4f  __attribute__((ext_vector_type(4)));

constexpr int CC  = 768;    // channels
constexpr int TT  = 1024;   // sequence length
constexpr int NH  = 12;     // heads
constexpr int HD  = 64;     // head dim
constexpr int NB  = 4;      // batch
constexpr int WIN = 4;      // relative window

// ---------------------------------------------------------------------------
// WMMA helpers (layouts per CDNA5 ISA 7.12.2, wave32)
// ---------------------------------------------------------------------------
__device__ __forceinline__ v8f zero8() {
  v8f z;
#pragma unroll
  for (int i = 0; i < 8; ++i) z[i] = 0.0f;
  return z;
}

__device__ __forceinline__ v8f wmma_f16(v16h a, v16h b, v8f c) {
  return __builtin_amdgcn_wmma_f32_16x16x32_f16(
      false, a, false, b, (short)0, c, false, false);
}

// A fragment (16x32, f16): lane&15 = M row; K split by lane half, 2 halves/VGPR.
__device__ __forceinline__ v16h load_a_frag(const _Float16* p, int stride,
                                            int rowBase, int kOff, int lane) {
  const int m = lane & 15;
  const int g = (lane >> 4) & 1;
  v16h a;
#pragma unroll
  for (int j = 0; j < 16; ++j) {
    const int kk = ((j >> 3) << 4) + (g << 3) + (j & 7);
    a[j] = p[(rowBase + m) * stride + kOff + kk];
  }
  return a;
}

// B fragment (32x16, f16), B stored row-major [k][n]: lane&15 = N col; K split by lane half like A.
__device__ __forceinline__ v16h load_b_row(const _Float16* p, int stride,
                                           int kOff, int colBase, int lane) {
  const int n = lane & 15;
  const int g = (lane >> 4) & 1;
  v16h b;
#pragma unroll
  for (int j = 0; j < 16; ++j) {
    const int kk = ((j >> 3) << 4) + (g << 3) + (j & 7);
    b[j] = p[(kOff + kk) * stride + colBase + n];
  }
  return b;
}

// pack 8 f32 -> 8 f16
__device__ __forceinline__ v8h cvt8(v4f a, v4f b) {
  v8h h;
#pragma unroll
  for (int i = 0; i < 4; ++i) {
    h[i] = (_Float16)a[i];
    h[4 + i] = (_Float16)b[i];
  }
  return h;
}

// async global -> LDS copy of one 16-byte chunk (GLOBAL_LOAD_ASYNC_TO_LDS_B128,
// GVS mode: mem = SADDR + VADDR32; LDS dest offset in VDST VGPR).
__device__ __forceinline__ void async_copy_b128(unsigned ldsOff, unsigned gOff,
                                                const void* base) {
  asm volatile("global_load_async_to_lds_b128 %0, %1, %2"
               :
               : "v"(ldsOff), "v"(gOff), "s"(base)
               : "memory");
}
__device__ __forceinline__ void async_wait0() {
  asm volatile("s_wait_asynccnt 0x0" ::: "memory");
}
// flat pointer -> LDS byte offset (LDS aperture keeps offset in addr[31:0])
__device__ __forceinline__ unsigned lds_off(const void* p) {
  return (unsigned)(uintptr_t)p;
}

// ---------------------------------------------------------------------------
// Kernel 1: fused QKV projection.  Y = W(768x768) * X_b(768x1024) (+bias),
// f16 head-split (b, h, t, d) out.  Q scaled by 1/sqrt(d).
// grid: (TT/128, CC/128, 3*NB), block 256 (8 waves, 4(M) x 2(N) wave grid)
// ---------------------------------------------------------------------------
__global__ __launch_bounds__(256) void proj_qkv_kernel(
    const float* __restrict__ x,
    const float* __restrict__ wq, const float* __restrict__ bq,
    const float* __restrict__ wk, const float* __restrict__ bk,
    const float* __restrict__ wv, const float* __restrict__ bv,
    _Float16* __restrict__ qf, _Float16* __restrict__ kf,
    _Float16* __restrict__ vf) {
  __shared__ __align__(16) _Float16 sA[128 * 32];   // W tile (M x K)
  __shared__ __align__(16) _Float16 sB[32 * 128];   // X tile (K x N)

  const int tid  = threadIdx.x;
  const int lane = tid & 31;
  const int wave = tid >> 5;
  const int wm   = wave & 3;
  const int wn   = wave >> 2;

  const int t0 = blockIdx.x * 128;
  const int m0 = blockIdx.y * 128;
  const int b  = blockIdx.z & 3;
  const int p  = blockIdx.z >> 2;   // 0=q 1=k 2=v

  const float*  W    = (p == 0) ? wq : (p == 1) ? wk : wv;
  const float*  bias = (p == 0) ? bq : (p == 1) ? bk : bv;
  _Float16*     out  = (p == 0) ? qf : (p == 1) ? kf : vf;
  const float   scale = (p == 0) ? 0.125f : 1.0f;

  // per-thread staging coordinates (16 contiguous elements each)
  const int arow = tid >> 1, ahalf = (tid & 1) * 16;
  const int brow = tid >> 3, bseg  = (tid & 7) * 16;

  v8f acc[2][4];
#pragma unroll
  for (int mt = 0; mt < 2; ++mt)
#pragma unroll
    for (int nt = 0; nt < 4; ++nt) acc[mt][nt] = zero8();

  for (int k0 = 0; k0 < CC; k0 += 32) {
    __syncthreads();
    {
      const float* src = W + (size_t)(m0 + arow) * CC + k0 + ahalf;
      const v4f f0 = *(const v4f*)(src + 0);
      const v4f f1 = *(const v4f*)(src + 4);
      const v4f f2 = *(const v4f*)(src + 8);
      const v4f f3 = *(const v4f*)(src + 12);
      *(v8h*)&sA[arow * 32 + ahalf + 0] = cvt8(f0, f1);
      *(v8h*)&sA[arow * 32 + ahalf + 8] = cvt8(f2, f3);
      __builtin_prefetch(src + 32, 0, 0);   // next K tile
    }
    {
      const float* src = x + ((size_t)b * CC + k0 + brow) * TT + t0 + bseg;
      const v4f f0 = *(const v4f*)(src + 0);
      const v4f f1 = *(const v4f*)(src + 4);
      const v4f f2 = *(const v4f*)(src + 8);
      const v4f f3 = *(const v4f*)(src + 12);
      *(v8h*)&sB[brow * 128 + bseg + 0] = cvt8(f0, f1);
      *(v8h*)&sB[brow * 128 + bseg + 8] = cvt8(f2, f3);
      __builtin_prefetch(src + 32 * TT, 0, 0);
    }
    __syncthreads();

    v16h afr[2];
#pragma unroll
    for (int mt = 0; mt < 2; ++mt)
      afr[mt] = load_a_frag(sA, 32, wm * 32 + mt * 16, 0, lane);
#pragma unroll
    for (int nt = 0; nt < 4; ++nt) {
      v16h bfr = load_b_row(sB, 128, 0, wn * 64 + nt * 16, lane);
#pragma unroll
      for (int mt = 0; mt < 2; ++mt)
        acc[mt][nt] = wmma_f16(afr[mt], bfr, acc[mt][nt]);
    }
  }

  const int g = (lane >> 4) & 1, n16 = lane & 15;
#pragma unroll
  for (int mt = 0; mt < 2; ++mt)
#pragma unroll
    for (int nt = 0; nt < 4; ++nt)
#pragma unroll
      for (int r = 0; r < 8; ++r) {
        const int o = m0 + wm * 32 + mt * 16 + r + g * 8;
        const int t = t0 + wn * 64 + nt * 16 + n16;
        const float v = (acc[mt][nt][r] + bias[o]) * scale;
        const int h = o >> 6, dd = o & 63;
        out[(((size_t)b * NH + h) * TT + t) * HD + dd] = (_Float16)v;
      }
}

// ---------------------------------------------------------------------------
// Kernel 2: attention with banded relative embeddings.
// One block per (b,h) x 128-row i-tile; two passes over 64-wide j-tiles.
// Q/V tiles stream via async global->LDS; K staged transposed so score
// B-fragments are contiguous.  grid: (TT/128, NB*NH)
// ---------------------------------------------------------------------------
__device__ __forceinline__ void score_tile(const _Float16* sQ,
                                           const _Float16* sKt,
                                           const float* rq, int i0, int j0,
                                           int wm, int wn, int lane,
                                           v8f s[2][2]) {
#pragma unroll
  for (int mt = 0; mt < 2; ++mt)
#pragma unroll
    for (int nt = 0; nt < 2; ++nt) s[mt][nt] = zero8();

#pragma unroll
  for (int ks = 0; ks < 2; ++ks) {     // K(=d) = 64 -> two WMMA steps
    const int kOff = ks * 32;
    v16h aq0 = load_a_frag(sQ, HD, wm * 32, kOff, lane);
    v16h aq1 = load_a_frag(sQ, HD, wm * 32 + 16, kOff, lane);
#pragma unroll
    for (int nt = 0; nt < 2; ++nt) {
      // B[k][n] = K^T[d][t] stored row-major in sKt -> contiguous loads
      v16h bk = load_b_row(sKt, 64, kOff, wn * 32 + nt * 16, lane);
      s[0][nt] = wmma_f16(aq0, bk, s[0][nt]);
      s[1][nt] = wmma_f16(aq1, bk, s[1][nt]);
    }
  }
  const int g = (lane >> 4) & 1, n16 = lane & 15;
#pragma unroll
  for (int mt = 0; mt < 2; ++mt)
#pragma unroll
    for (int nt = 0; nt < 2; ++nt)
#pragma unroll
      for (int r = 0; r < 8; ++r) {
        const int il  = wm * 32 + mt * 16 + r + g * 8;
        const int rel = (j0 + wn * 32 + nt * 16 + n16) - (i0 + il);
        if (rel >= -WIN && rel <= WIN) s[mt][nt][r] += rq[il * 10 + rel + WIN];
      }
}

__global__ __launch_bounds__(256) void attn_kernel(
    const _Float16* __restrict__ qf, const _Float16* __restrict__ kf,
    const _Float16* __restrict__ vf, const float* __restrict__ embk,
    const float* __restrict__ embv, _Float16* __restrict__ aof) {
  __shared__ __align__(16) char smem[61440];
  _Float16* sQ     = (_Float16*)(smem);           // 128x64 f16  (16 KB)
  _Float16* sKt    = (_Float16*)(smem + 16384);   //  64(d)x64(t) f16 (8 KB)
  _Float16* sV     = (_Float16*)(smem + 24576);   //  64x64 f16  ( 8 KB)
  _Float16* sP     = (_Float16*)(smem + 32768);   // 128x64 f16  (16 KB)
  float*    rq     = (float*)(smem + 49152);      // 128x10 f32
  float*    pband  = (float*)(smem + 54272);      // 128x10 f32
  float*    rowmax = (float*)(smem + 59392);      // 128 f32
  float*    rowred = (float*)(smem + 59904);      // 2x128 f32
  float*    rowsum = (float*)(smem + 60928);      // 128 f32
  float*    oxch   = (float*)(smem);              // epilogue reuse: 128x64 f32

  const int tid  = threadIdx.x;
  const int lane = tid & 31;
  const int wave = tid >> 5;
  const int wm   = wave & 3;
  const int wn   = wave >> 2;
  const int g    = (lane >> 4) & 1;
  const int n16  = lane & 15;

  const int bh = blockIdx.y;
  const int b  = bh / NH, h = bh % NH;
  const size_t base = (size_t)bh * TT * HD;
  const int i0 = blockIdx.x * 128;

  // ---- async-stream Q row-tile (16 KB = 1024 x b128 chunks) ----
  {
    const _Float16* qsrc = qf + base + (size_t)i0 * HD;
    const unsigned ldsQ = lds_off(sQ);
#pragma unroll
    for (int i = 0; i < 4; ++i) {
      const unsigned c = (unsigned)(i * 256 + tid) * 16u;
      async_copy_b128(ldsQ + c, c, qsrc);
    }
    async_wait0();
  }
  __syncthreads();

  // rq[i][r] = q_i . emb_rel_k[r]
  for (int l = tid; l < 128 * 9; l += 256) {
    const int i = l / 9, r = l % 9;
    float s = 0.0f;
#pragma unroll 8
    for (int dd = 0; dd < HD; ++dd)
      s += (float)sQ[i * HD + dd] * embk[r * HD + dd];
    rq[i * 10 + r] = s;
  }
  for (int l = tid; l < 128 * 10; l += 256) pband[l] = 0.0f;
  __syncthreads();

  // per-thread K staging coords: chunk of 8 contiguous halves, transposed store
  const int kc_t0 = tid >> 3;            // t row for chunk 0 (chunk1: +32)
  const int kc_d  = (tid & 7) * 8;       // d start

  // ---------------- pass 1: exact row max ----------------
  float pmax[16];
#pragma unroll
  for (int sl = 0; sl < 16; ++sl) pmax[sl] = -3.0e38f;

  for (int jt = 0; jt < 16; ++jt) {
    const int j0 = jt * 64;
    __syncthreads();
    const _Float16* ksrc = kf + base + (size_t)j0 * HD;
#pragma unroll
    for (int i = 0; i < 2; ++i) {
      const int t = kc_t0 + i * 32;
      const v8h kv = *(const v8h*)(ksrc + t * HD + kc_d);
#pragma unroll
      for (int e = 0; e < 8; ++e) sKt[(kc_d + e) * 64 + t] = kv[e];
    }
    __builtin_prefetch(ksrc + 64 * HD + tid * 8, 0, 0);
    __syncthreads();

    v8f s[2][2];
    score_tile(sQ, sKt, rq, i0, j0, wm, wn, lane, s);
#pragma unroll
    for (int mt = 0; mt < 2; ++mt)
#pragma unroll
      for (int nt = 0; nt < 2; ++nt)
#pragma unroll
        for (int r = 0; r < 8; ++r)
          pmax[mt * 8 + r] = fmaxf(pmax[mt * 8 + r], s[mt][nt][r]);
  }
#pragma unroll
  for (int sl = 0; sl < 16; ++sl) {
    float v = pmax[sl];
    v = fmaxf(v, __shfl_xor(v, 8, 16));
    v = fmaxf(v, __shfl_xor(v, 4, 16));
    v = fmaxf(v, __shfl_xor(v, 2, 16));
    v = fmaxf(v, __shfl_xor(v, 1, 16));
    pmax[sl] = v;
  }
  if (n16 == 0) {
#pragma unroll
    for (int sl = 0; sl < 16; ++sl) {
      const int il = wm * 32 + (sl >> 3) * 16 + (sl & 7) + g * 8;
      rowred[wn * 128 + il] = pmax[sl];
    }
  }
  __syncthreads();
  for (int l = tid; l < 128; l += 256)
    rowmax[l] = fmaxf(rowred[l], rowred[128 + l]);
  __syncthreads();

  // ---------------- pass 2: exp, row sums, P*V ----------------
  float psum[16];
#pragma unroll
  for (int sl = 0; sl < 16; ++sl) psum[sl] = 0.0f;
  v8f oacc[2][4];
#pragma unroll
  for (int mt = 0; mt < 2; ++mt)
#pragma unroll
    for (int dt = 0; dt < 4; ++dt) oacc[mt][dt] = zero8();

  for (int jt = 0; jt < 16; ++jt) {
    const int j0 = jt * 64;
    __syncthreads();
    // V tile: async global -> LDS (row-major is already fragment-friendly)
    {
      const _Float16* vsrc = vf + base + (size_t)j0 * HD;
      const unsigned ldsV = lds_off(sV);
#pragma unroll
      for (int i = 0; i < 2; ++i) {
        const unsigned c = (unsigned)(i * 256 + tid) * 16u;
        async_copy_b128(ldsV + c, c, vsrc);
      }
    }
    // K tile: staged transposed
    const _Float16* ksrc = kf + base + (size_t)j0 * HD;
#pragma unroll
    for (int i = 0; i < 2; ++i) {
      const int t = kc_t0 + i * 32;
      const v8h kv = *(const v8h*)(ksrc + t * HD + kc_d);
#pragma unroll
      for (int e = 0; e < 8; ++e) sKt[(kc_d + e) * 64 + t] = kv[e];
    }
    async_wait0();
    __syncthreads();

    v8f s[2][2];
    score_tile(sQ, sKt, rq, i0, j0, wm, wn, lane, s);

#pragma unroll
    for (int mt = 0; mt < 2; ++mt)
#pragma unroll
      for (int nt = 0; nt < 2; ++nt)
#pragma unroll
        for (int r = 0; r < 8; ++r) {
          const int il = wm * 32 + mt * 16 + r + g * 8;
          const int jl = wn * 32 + nt * 16 + n16;
          const float p = __expf(s[mt][nt][r] - rowmax[il]);
          psum[mt * 8 + r] += p;
          const int rel = (j0 + jl) - (i0 + il);
          if (rel >= -WIN && rel <= WIN) pband[il * 10 + rel + WIN] = p;
          sP[il * 64 + jl] = (_Float16)p;
        }
    __syncthreads();

    v16h ap[2];
#pragma unroll
    for (int mt = 0; mt < 2; ++mt)
      ap[mt] = load_a_frag(sP, 64, wm * 32 + mt * 16, wn * 32, lane);
#pragma unroll
    for (int dt = 0; dt < 4; ++dt) {
      v16h bv = load_b_row(sV, HD, wn * 32, dt * 16, lane);
#pragma unroll
      for (int mt = 0; mt < 2; ++mt)
        oacc[mt][dt] = wmma_f16(ap[mt], bv, oacc[mt][dt]);
    }
  }

#pragma unroll
  for (int sl = 0; sl < 16; ++sl) {
    float v = psum[sl];
    v += __shfl_xor(v, 8, 16);
    v += __shfl_xor(v, 4, 16);
    v += __shfl_xor(v, 2, 16);
    v += __shfl_xor(v, 1, 16);
    psum[sl] = v;
  }
  if (n16 == 0) {
#pragma unroll
    for (int sl = 0; sl < 16; ++sl) {
      const int il = wm * 32 + (sl >> 3) * 16 + (sl & 7) + g * 8;
      rowred[wn * 128 + il] = psum[sl];
    }
  }
  __syncthreads();
  for (int l = tid; l < 128; l += 256)
    rowsum[l] = rowred[l] + rowred[128 + l];
  __syncthreads();

  if (wn == 1) {
#pragma unroll
    for (int mt = 0; mt < 2; ++mt)
#pragma unroll
      for (int dt = 0; dt < 4; ++dt)
#pragma unroll
        for (int r = 0; r < 8; ++r) {
          const int il = wm * 32 + mt * 16 + r + g * 8;
          const int d = dt * 16 + n16;
          oxch[il * HD + d] = oacc[mt][dt][r];
        }
  }
  __syncthreads();
  if (wn == 0) {
#pragma unroll
    for (int mt = 0; mt < 2; ++mt)
#pragma unroll
      for (int dt = 0; dt < 4; ++dt)
#pragma unroll
        for (int r = 0; r < 8; ++r) {
          const int il = wm * 32 + mt * 16 + r + g * 8;
          const int d = dt * 16 + n16;
          float val = oacc[mt][dt][r] + oxch[il * HD + d];
          float rv = 0.0f;
#pragma unroll
          for (int r2 = 0; r2 < 9; ++r2)
            rv += pband[il * 10 + r2] * embv[r2 * HD + d];
          val = (val + rv) / rowsum[il];
          aof[((size_t)b * CC + h * HD + d) * TT + (i0 + il)] = (_Float16)val;
        }
  }
}

// ---------------------------------------------------------------------------
// Kernel 3: output projection.  Y = W_o * AO_b + b_o, f32 out (b, c, t).
// grid: (TT/128, CC/128, NB), block 256
// ---------------------------------------------------------------------------
__global__ __launch_bounds__(256) void out_proj_kernel(
    const _Float16* __restrict__ ao, const float* __restrict__ wo,
    const float* __restrict__ bo, float* __restrict__ out) {
  __shared__ __align__(16) _Float16 sA[128 * 32];
  __shared__ __align__(16) _Float16 sB[32 * 128];

  const int tid  = threadIdx.x;
  const int lane = tid & 31;
  const int wave = tid >> 5;
  const int wm   = wave & 3;
  const int wn   = wave >> 2;

  const int t0 = blockIdx.x * 128;
  const int m0 = blockIdx.y * 128;
  const int b  = blockIdx.z;

  const int arow = tid >> 1, ahalf = (tid & 1) * 16;
  const int brow = tid >> 3, bseg  = (tid & 7) * 16;

  v8f acc[2][4];
#pragma unroll
  for (int mt = 0; mt < 2; ++mt)
#pragma unroll
    for (int nt = 0; nt < 4; ++nt) acc[mt][nt] = zero8();

  for (int k0 = 0; k0 < CC; k0 += 32) {
    __syncthreads();
    {
      const float* src = wo + (size_t)(m0 + arow) * CC + k0 + ahalf;
      const v4f f0 = *(const v4f*)(src + 0);
      const v4f f1 = *(const v4f*)(src + 4);
      const v4f f2 = *(const v4f*)(src + 8);
      const v4f f3 = *(const v4f*)(src + 12);
      *(v8h*)&sA[arow * 32 + ahalf + 0] = cvt8(f0, f1);
      *(v8h*)&sA[arow * 32 + ahalf + 8] = cvt8(f2, f3);
      __builtin_prefetch(src + 32, 0, 0);
    }
    {
      const _Float16* src = ao + ((size_t)b * CC + k0 + brow) * TT + t0 + bseg;
      *(v8h*)&sB[brow * 128 + bseg + 0] = *(const v8h*)(src + 0);
      *(v8h*)&sB[brow * 128 + bseg + 8] = *(const v8h*)(src + 8);
      __builtin_prefetch(src + 32 * TT, 0, 0);
    }
    __syncthreads();

    v16h afr[2];
#pragma unroll
    for (int mt = 0; mt < 2; ++mt)
      afr[mt] = load_a_frag(sA, 32, wm * 32 + mt * 16, 0, lane);
#pragma unroll
    for (int nt = 0; nt < 4; ++nt) {
      v16h bfr = load_b_row(sB, 128, 0, wn * 64 + nt * 16, lane);
#pragma unroll
      for (int mt = 0; mt < 2; ++mt)
        acc[mt][nt] = wmma_f16(afr[mt], bfr, acc[mt][nt]);
    }
  }

  const int g = (lane >> 4) & 1, n16 = lane & 15;
#pragma unroll
  for (int mt = 0; mt < 2; ++mt)
#pragma unroll
    for (int nt = 0; nt < 4; ++nt)
#pragma unroll
      for (int r = 0; r < 8; ++r) {
        const int o = m0 + wm * 32 + mt * 16 + r + g * 8;
        const int t = t0 + wn * 64 + nt * 16 + n16;
        out[((size_t)b * CC + o) * TT + t] = acc[mt][nt][r] + bo[o];
      }
}

// ---------------------------------------------------------------------------
extern "C" void kernel_launch(void* const* d_in, const int* in_sizes, int n_in,
                              void* d_out, int out_size, void* d_ws,
                              size_t ws_size, hipStream_t stream) {
  const float* x    = (const float*)d_in[0];
  const float* wq   = (const float*)d_in[1];
  const float* bq   = (const float*)d_in[2];
  const float* wk   = (const float*)d_in[3];
  const float* bk   = (const float*)d_in[4];
  const float* wv   = (const float*)d_in[5];
  const float* bv   = (const float*)d_in[6];
  const float* wo   = (const float*)d_in[7];
  const float* bo   = (const float*)d_in[8];
  const float* embk = (const float*)d_in[9];
  const float* embv = (const float*)d_in[10];
  float* out = (float*)d_out;

  const size_t QS = (size_t)NB * NH * TT * HD;   // == NB*CC*TT
  _Float16* ws  = (_Float16*)d_ws;
  _Float16* qf  = ws;
  _Float16* kf  = ws + QS;
  _Float16* vf  = ws + 2 * QS;
  _Float16* aof = ws + 3 * QS;

  dim3 blk(256);
  proj_qkv_kernel<<<dim3(TT / 128, CC / 128, 3 * NB), blk, 0, stream>>>(
      x, wq, bq, wk, bk, wv, bv, qf, kf, vf);
  attn_kernel<<<dim3(TT / 128, NB * NH), blk, 0, stream>>>(qf, kf, vf, embk,
                                                           embv, aof);
  out_proj_kernel<<<dim3(TT / 128, CC / 128, NB), blk, 0, stream>>>(aof, wo,
                                                                    bo, out);
}
